// MLAAttention_66486093742226
// MI455X (gfx1250) — compile-verified, matrix operand order checked by reading, not run on hardware
//
#include <hip/hip_runtime.h>

// ---------------------------------------------------------------- types
typedef __bf16 bf16;
typedef __attribute__((ext_vector_type(8)))  __bf16 v8bf;
typedef __attribute__((ext_vector_type(16))) __bf16 v16bf;
typedef __attribute__((ext_vector_type(8)))  float  v8f;
typedef int v4i_vs __attribute__((vector_size(16)));   // matches builtin proto V4i

__device__ __forceinline__ bf16 f2bf(float f) {
  unsigned u = __builtin_bit_cast(unsigned, f);
  u += 0x7FFFu + ((u >> 16) & 1u);            // round-to-nearest-even
  unsigned short hsh = (unsigned short)(u >> 16);
  return __builtin_bit_cast(bf16, hsh);
}

// ISA 16-bit fragment: lane<16 holds K {base..base+7, base+16..base+23} of its row
__device__ __forceinline__ v16bf ldfrag(const bf16* p) {
  v8bf lo = *reinterpret_cast<const v8bf*>(p);
  v8bf hi = *reinterpret_cast<const v8bf*>(p + 16);
  return __builtin_shufflevector(lo, hi, 0,1,2,3,4,5,6,7,8,9,10,11,12,13,14,15);
}

__device__ __forceinline__ v8f wmma_bf16(v16bf a, v16bf b, v8f c) {
  return __builtin_amdgcn_wmma_f32_16x16x32_bf16(false, a, false, b, (short)0, c, false, false);
}

// ---------------------------------------------------------------- CDNA5 async copy (ASYNCcnt path), guarded
#if __has_builtin(__builtin_amdgcn_global_load_async_to_lds_b128)
#define HAVE_ASYNC 1
#else
#define HAVE_ASYNC 0
#endif

__device__ __forceinline__ void async_cp16(const bf16* g, bf16* l) {
#if HAVE_ASYNC
  __builtin_amdgcn_global_load_async_to_lds_b128(
      (__attribute__((address_space(1))) v4i_vs*)g,
      (__attribute__((address_space(3))) v4i_vs*)l, 0, 0);
#else
  *reinterpret_cast<v8bf*>(l) = *reinterpret_cast<const v8bf*>(g);
#endif
}

__device__ __forceinline__ void async_wait0() {
#if HAVE_ASYNC
#if __has_builtin(__builtin_amdgcn_s_wait_asynccnt)
  __builtin_amdgcn_s_wait_asynccnt(0);
#else
  asm volatile("s_wait_asynccnt 0" ::: "memory");
#endif
#endif
}

__device__ __forceinline__ float wave_sum32(float v) {
  #pragma unroll
  for (int m = 16; m >= 1; m >>= 1) v += __shfl_xor(v, m, 32);
  return v;
}
// reductions within 16-lane halves (masks keep bit 4 fixed)
__device__ __forceinline__ float half_max16(float v) {
  #pragma unroll
  for (int m = 8; m >= 1; m >>= 1) v = fmaxf(v, __shfl_xor(v, m, 32));
  return v;
}
__device__ __forceinline__ float half_sum16(float v) {
  #pragma unroll
  for (int m = 8; m >= 1; m >>= 1) v += __shfl_xor(v, m, 32);
  return v;
}

// ---------------------------------------------------------------- convert
__global__ void cvt_f32_bf16(const float* __restrict__ in, bf16* __restrict__ out, int n) {
  int i = blockIdx.x * blockDim.x + threadIdx.x;
  int stride = gridDim.x * blockDim.x;
  for (; i < n; i += stride) out[i] = f2bf(in[i]);
}

// ---------------------------------------------------------------- GEMM: C[M,N] = A[M,K] * B[N,K]^T (bf16 in, f32 out)
// block 256 threads = 8 waves; tile 128x256x32; wave = 64x64 (4x4 WMMA accs)
// double-buffered LDS fed by async global->LDS copies
#define GP 40
__global__ __launch_bounds__(256) void gemm_bt_f32out(
    const bf16* __restrict__ A, const bf16* __restrict__ B, float* __restrict__ C,
    int M, int N, int K) {
  __shared__ bf16 As[2][128 * GP];
  __shared__ bf16 Bs[2][256 * GP];
  const int tid  = threadIdx.x;
  const int wave = tid >> 5, lane = tid & 31;
  const int hi = lane >> 4, r = lane & 15;
  const int bm = blockIdx.y * 128, bn = blockIdx.x * 256;
  const int wm = (wave >> 2) * 64;     // 0,64
  const int wn = (wave & 3) * 64;      // 0,64,128,192

  v8f acc[4][4];
  v8f zero = {0.f,0.f,0.f,0.f,0.f,0.f,0.f,0.f};
  #pragma unroll
  for (int i = 0; i < 4; ++i)
    #pragma unroll
    for (int j = 0; j < 4; ++j) acc[i][j] = zero;

  auto issue = [&](int buf, int kt) {
    // A tile: 128x32 = 512 chunks of 8 bf16 -> 2 per thread
    #pragma unroll
    for (int i = 0; i < 2; ++i) {
      int c = tid + 256 * i;
      int row = c >> 2, col = (c & 3) * 8;
      async_cp16(A + (size_t)(bm + row) * K + kt + col, &As[buf][row * GP + col]);
    }
    // B tile: 256x32 = 1024 chunks -> 4 per thread
    #pragma unroll
    for (int i = 0; i < 4; ++i) {
      int c = tid + 256 * i;
      int row = c >> 2, col = (c & 3) * 8;
      async_cp16(B + (size_t)(bn + row) * K + kt + col, &Bs[buf][row * GP + col]);
    }
  };

  issue(0, 0);
  async_wait0();
  __syncthreads();

  const int nk = K >> 5;
  for (int t = 0; t < nk; ++t) {
    const int cur = t & 1, nxt = cur ^ 1;
    if (t + 1 < nk) issue(nxt, (t + 1) << 5);   // DMA next tile while we compute

    v16bf af[4], bfr[4];
    #pragma unroll
    for (int i = 0; i < 4; ++i) af[i]  = ldfrag(&As[cur][(wm + i * 16 + r) * GP + hi * 8]);
    #pragma unroll
    for (int j = 0; j < 4; ++j) bfr[j] = ldfrag(&Bs[cur][(wn + j * 16 + r) * GP + hi * 8]);
    #pragma unroll
    for (int i = 0; i < 4; ++i)
      #pragma unroll
      for (int j = 0; j < 4; ++j)
        acc[i][j] = wmma_bf16(af[i], bfr[j], acc[i][j]);

    async_wait0();
    __syncthreads();
  }

  #pragma unroll
  for (int i = 0; i < 4; ++i)
    #pragma unroll
    for (int j = 0; j < 4; ++j) {
      int m0 = bm + wm + i * 16 + 8 * hi;  // C-frag: lane holds col n=r, rows m0..m0+7
      int n0 = bn + wn + j * 16 + r;
      #pragma unroll
      for (int rr = 0; rr < 8; ++rr)
        C[(size_t)(m0 + rr) * N + n0] = acc[i][j][rr];
    }
}

// ---------------------------------------------------------------- c_kv RMS (rows of 512), one wave per row
__global__ __launch_bounds__(256) void rms_rows512(const float* __restrict__ Y,
    const float* __restrict__ g, bf16* __restrict__ out) {
  int wave = threadIdx.x >> 5, lane = threadIdx.x & 31;
  int row = blockIdx.x * 8 + wave;
  const float* y = Y + (size_t)row * 512;
  float vals[16];
  float s = 0.f;
  #pragma unroll
  for (int j = 0; j < 16; ++j) { vals[j] = y[lane + 32 * j]; s += vals[j] * vals[j]; }
  s = wave_sum32(s);
  float inv = rsqrtf(s * (1.0f / 512.0f) + 1e-6f);
  bf16* o = out + (size_t)row * 512;
  #pragma unroll
  for (int j = 0; j < 16; ++j) o[lane + 32 * j] = f2bf(vals[j] * inv * g[lane + 32 * j]);
}

// ---------------------------------------------------------------- Q finalize: per (s,h) RMS(nope), RMS+RoPE(rope), *1/sqrt(HD)
__global__ __launch_bounds__(256) void q_finalize(const float* __restrict__ Yn,
    const float* __restrict__ Yr, const float* __restrict__ gn, const float* __restrict__ gr,
    const float* __restrict__ cosp, const float* __restrict__ sinp, bf16* __restrict__ Q) {
  int wave = threadIdx.x >> 5, lane = threadIdx.x & 31;
  int idx = blockIdx.x * 8 + wave;
  int s = idx >> 4, h = idx & 15;
  const float SC = 0.08838834764831845f;   // 1/sqrt(128) folded into Q
  float a0 = Yn[(size_t)s * 1024 + h * 64 + lane];
  float a1 = Yn[(size_t)s * 1024 + h * 64 + lane + 32];
  float ss = wave_sum32(a0 * a0 + a1 * a1);
  float inv = rsqrtf(ss * (1.0f / 64.0f) + 1e-6f);
  bf16* q = Q + (size_t)s * 2048 + h * 128;
  q[lane]      = f2bf(a0 * inv * gn[lane] * SC);
  q[lane + 32] = f2bf(a1 * inv * gn[lane + 32] * SC);
  float b0 = Yr[(size_t)s * 1024 + h * 64 + lane];
  float b1 = Yr[(size_t)s * 1024 + h * 64 + lane + 32];
  float s2 = wave_sum32(b0 * b0 + b1 * b1);
  float inv2 = rsqrtf(s2 * (1.0f / 64.0f) + 1e-6f);
  float b0n = b0 * inv2 * gr[lane];
  float b1n = b1 * inv2 * gr[lane + 32];
  float c0 = cosp[(size_t)s * 64 + lane],      s0 = sinp[(size_t)s * 64 + lane];
  float c1 = cosp[(size_t)s * 64 + lane + 32], s1 = sinp[(size_t)s * 64 + lane + 32];
  q[64 + lane]      = f2bf((b0n * c0 - b1n * s0) * SC);
  q[64 + lane + 32] = f2bf((b1n * c1 + b0n * s1) * SC);
}

// ---------------------------------------------------------------- K/V finalize: RoPE on k_rope; V stored transposed [kv*128+d][S]
__global__ __launch_bounds__(256) void kv_finalize(const float* __restrict__ Kn,
    const float* __restrict__ Kr, const float* __restrict__ Vf,
    const float* __restrict__ cosp, const float* __restrict__ sinp,
    bf16* __restrict__ K, bf16* __restrict__ Vt) {
  int wave = threadIdx.x >> 5, lane = threadIdx.x & 31;
  int idx = blockIdx.x * 8 + wave;
  int s = idx >> 2, kv = idx & 3;
  bf16* k = K + (size_t)s * 512 + kv * 128;
  float a0 = Kn[(size_t)s * 256 + kv * 64 + lane];
  float a1 = Kn[(size_t)s * 256 + kv * 64 + lane + 32];
  k[lane] = f2bf(a0); k[lane + 32] = f2bf(a1);
  float b0 = Kr[(size_t)s * 256 + kv * 64 + lane];
  float b1 = Kr[(size_t)s * 256 + kv * 64 + lane + 32];
  float c0 = cosp[(size_t)s * 64 + lane],      s0 = sinp[(size_t)s * 64 + lane];
  float c1 = cosp[(size_t)s * 64 + lane + 32], s1 = sinp[(size_t)s * 64 + lane + 32];
  k[64 + lane]      = f2bf(b0 * c0 - b1 * s0);
  k[64 + lane + 32] = f2bf(b1 * c1 + b0 * s1);
  #pragma unroll
  for (int j = 0; j < 4; ++j) {
    int d = lane + 32 * j;
    Vt[((size_t)kv * 128 + d) * 2048 + s] = f2bf(Vf[(size_t)s * 512 + kv * 128 + d]);
  }
}

// ---------------------------------------------------------------- flash attention (causal), double-buffered async K/V tiles
// block = (64-query tile, head); 4 waves x 16 queries each; keys in chunks of 32
#define KP 136
#define VP 40
__global__ __launch_bounds__(128) void mla_attn(const bf16* __restrict__ Q,
    const bf16* __restrict__ K, const bf16* __restrict__ Vt, bf16* __restrict__ O) {
  __shared__ bf16 Kt[2][32 * KP];
  __shared__ bf16 Vs[2][128 * VP];
  __shared__ bf16 Pw[4][16 * VP];
  const int tid = threadIdx.x;
  const int wave = tid >> 5, lane = tid & 31;
  const int hi = lane >> 4, r = lane & 15;
  const int h  = blockIdx.y;
  const int kv = h >> 2;                  // n_rep = NH/NKV = 4
  const int q0b = blockIdx.x * 64;
  const int q0  = q0b + wave * 16;

  // Q fragments for this wave's 16 rows (1/sqrt(HD) pre-folded)
  v16bf qf[4];
  {
    const bf16* qrow = Q + (size_t)(q0 + r) * 2048 + h * 128 + hi * 8;
    #pragma unroll
    for (int c = 0; c < 4; ++c) qf[c] = ldfrag(qrow + c * 32);
  }

  float m_i[8], l_i[8];
  v8f Oa[8];
  v8f zero = {0.f,0.f,0.f,0.f,0.f,0.f,0.f,0.f};
  #pragma unroll
  for (int rr = 0; rr < 8; ++rr) { m_i[rr] = -1e30f; l_i[rr] = 0.f; }
  #pragma unroll
  for (int j = 0; j < 8; ++j) Oa[j] = zero;

  auto issue = [&](int buf, int k0) {
    // K tile: 32 keys x 128 dims = 512 chunks -> 4 per thread
    #pragma unroll
    for (int i = 0; i < 4; ++i) {
      int c = tid + 128 * i;
      int row = c >> 4, col = (c & 15) * 8;
      async_cp16(K + (size_t)(k0 + row) * 512 + kv * 128 + col, &Kt[buf][row * KP + col]);
    }
    // V^T tile: 128 dims x 32 keys (one row per thread)
    const bf16* src = Vt + ((size_t)kv * 128 + tid) * 2048 + k0;
    #pragma unroll
    for (int j = 0; j < 4; ++j)
      async_cp16(src + j * 8, &Vs[buf][tid * VP + j * 8]);
  };

  const int kend = q0b + 64;
  const int nkt = kend >> 5;              // >= 2
  issue(0, 0);
  async_wait0();
  __syncthreads();

  for (int t = 0; t < nkt; ++t) {
    const int cur = t & 1, nxt = cur ^ 1;
    const int k0 = t << 5;
    if (t + 1 < nkt) issue(nxt, k0 + 32);

    // scores: two 16x16 tiles over 4 HD chunks
    v8f sc[2];
    #pragma unroll
    for (int tt = 0; tt < 2; ++tt) {
      sc[tt] = zero;
      #pragma unroll
      for (int c = 0; c < 4; ++c) {
        v16bf kf = ldfrag(&Kt[cur][(tt * 16 + r) * KP + c * 32 + hi * 8]);
        sc[tt] = wmma_bf16(qf[c], kf, sc[tt]);
      }
    }
    // causal mask + online softmax (C-frag layout: lane col n=r, rows rr+8*hi)
    #pragma unroll
    for (int rr = 0; rr < 8; ++rr) {
      int qr = q0 + rr + 8 * hi;
      if (k0 + r > qr)      sc[0][rr] = -1e30f;
      if (k0 + 16 + r > qr) sc[1][rr] = -1e30f;
      float mx = half_max16(fmaxf(sc[0][rr], sc[1][rr]));
      float nm = fmaxf(m_i[rr], mx);
      float al = __expf(m_i[rr] - nm);
      m_i[rr] = nm;
      float p0 = __expf(sc[0][rr] - nm);
      float p1 = __expf(sc[1][rr] - nm);
      sc[0][rr] = p0; sc[1][rr] = p1;
      float rs = half_sum16(p0 + p1);
      l_i[rr] = l_i[rr] * al + rs;
      #pragma unroll
      for (int j = 0; j < 8; ++j) Oa[j][rr] *= al;
      int prow = (rr + 8 * hi) * VP;
      Pw[wave][prow + r]      = f2bf(p0);
      Pw[wave][prow + 16 + r] = f2bf(p1);
    }
    // restripe P (C-layout -> A-layout) via per-wave LDS (same-wave DS ops are in-order)
    v16bf pf = ldfrag(&Pw[wave][r * VP + hi * 8]);
    #pragma unroll
    for (int j = 0; j < 8; ++j) {
      v16bf vf = ldfrag(&Vs[cur][(j * 16 + r) * VP + hi * 8]);
      Oa[j] = wmma_bf16(pf, vf, Oa[j]);
    }
    async_wait0();
    __syncthreads();
  }
  #pragma unroll
  for (int j = 0; j < 8; ++j)
    #pragma unroll
    for (int rr = 0; rr < 8; ++rr) {
      int srow = q0 + rr + 8 * hi;
      O[(size_t)srow * 2048 + h * 128 + j * 16 + r] = f2bf(Oa[j][rr] / l_i[rr]);
    }
}

// ---------------------------------------------------------------- launch
extern "C" void kernel_launch(void* const* d_in, const int* in_sizes, int n_in,
                              void* d_out, int out_size, void* d_ws, size_t ws_size,
                              hipStream_t stream) {
  const float* x    = (const float*)d_in[0];
  const float* cosp = (const float*)d_in[1];
  const float* sinp = (const float*)d_in[2];
  const float* Wqn  = (const float*)d_in[3];   // [1024,2048]
  const float* Wqr  = (const float*)d_in[4];   // [1024,2048]
  const float* gqn  = (const float*)d_in[5];   // [64]
  const float* gqr  = (const float*)d_in[6];   // [64]
  const float* Wkvd = (const float*)d_in[7];   // [512,2048]
  const float* gckv = (const float*)d_in[8];   // [512]
  const float* Wkn  = (const float*)d_in[9];   // [256,512]
  const float* Wkr  = (const float*)d_in[10];  // [256,512]
  const float* Wv   = (const float*)d_in[11];  // [512,512]
  const float* Wo   = (const float*)d_in[12];  // [2048,2048]
  float* out = (float*)d_out;
  (void)in_sizes; (void)n_in; (void)out_size; (void)ws_size;

  char* p = (char*)d_ws;
  auto carve = [&](size_t bytes) { char* q = p; p += (bytes + 255) & ~(size_t)255; return q; };

  bf16* xb    = (bf16*)carve((size_t)2048 * 2048 * 2);
  bf16* WqnB  = (bf16*)carve((size_t)1024 * 2048 * 2);
  bf16* WqrB  = (bf16*)carve((size_t)1024 * 2048 * 2);
  bf16* WkvdB = (bf16*)carve((size_t)512 * 2048 * 2);
  bf16* WknB  = (bf16*)carve((size_t)256 * 512 * 2);
  bf16* WkrB  = (bf16*)carve((size_t)256 * 512 * 2);
  bf16* WvB   = (bf16*)carve((size_t)512 * 512 * 2);
  bf16* WoB   = (bf16*)carve((size_t)2048 * 2048 * 2);
  float* Yqn  = (float*)carve((size_t)2048 * 1024 * 4);
  float* Yqr  = (float*)carve((size_t)2048 * 1024 * 4);
  float* Yckv = (float*)carve((size_t)2048 * 512 * 4);
  bf16* ckvB  = (bf16*)carve((size_t)2048 * 512 * 2);
  float* Knf  = (float*)carve((size_t)2048 * 256 * 4);
  float* Krf  = (float*)carve((size_t)2048 * 256 * 4);
  float* Vff  = (float*)carve((size_t)2048 * 512 * 4);
  bf16* Qb    = (bf16*)carve((size_t)2048 * 2048 * 2);
  bf16* Kb    = (bf16*)carve((size_t)2048 * 512 * 2);
  bf16* Vtb   = (bf16*)carve((size_t)512 * 2048 * 2);
  bf16* attnB = (bf16*)carve((size_t)2048 * 2048 * 2);

  // f32 -> bf16 conversions
  cvt_f32_bf16<<<2048, 256, 0, stream>>>(x,    xb,    2048 * 2048);
  cvt_f32_bf16<<<1024, 256, 0, stream>>>(Wqn,  WqnB,  1024 * 2048);
  cvt_f32_bf16<<<1024, 256, 0, stream>>>(Wqr,  WqrB,  1024 * 2048);
  cvt_f32_bf16<<<512,  256, 0, stream>>>(Wkvd, WkvdB, 512 * 2048);
  cvt_f32_bf16<<<128,  256, 0, stream>>>(Wkn,  WknB,  256 * 512);
  cvt_f32_bf16<<<128,  256, 0, stream>>>(Wkr,  WkrB,  256 * 512);
  cvt_f32_bf16<<<256,  256, 0, stream>>>(Wv,   WvB,   512 * 512);
  cvt_f32_bf16<<<2048, 256, 0, stream>>>(Wo,   WoB,   2048 * 2048);

  // projections (C = A * W^T); block tile 128x256
  gemm_bt_f32out<<<dim3(4, 16), 256, 0, stream>>>(xb, WqnB,  Yqn,  2048, 1024, 2048);
  gemm_bt_f32out<<<dim3(4, 16), 256, 0, stream>>>(xb, WqrB,  Yqr,  2048, 1024, 2048);
  gemm_bt_f32out<<<dim3(2, 16), 256, 0, stream>>>(xb, WkvdB, Yckv, 2048, 512,  2048);

  rms_rows512<<<256, 256, 0, stream>>>(Yckv, gckv, ckvB);

  gemm_bt_f32out<<<dim3(1, 16), 256, 0, stream>>>(ckvB, WknB, Knf, 2048, 256, 512);
  gemm_bt_f32out<<<dim3(1, 16), 256, 0, stream>>>(ckvB, WkrB, Krf, 2048, 256, 512);
  gemm_bt_f32out<<<dim3(2, 16), 256, 0, stream>>>(ckvB, WvB,  Vff, 2048, 512, 512);

  q_finalize<<<4096, 256, 0, stream>>>(Yqn, Yqr, gqn, gqr, cosp, sinp, Qb);
  kv_finalize<<<1024, 256, 0, stream>>>(Knf, Krf, Vff, cosp, sinp, Kb, Vtb);

  // attention: grid (query tiles of 64, heads)
  mla_attn<<<dim3(32, 16), 128, 0, stream>>>(Qb, Kb, Vtb, attnB);

  // output projection straight into d_out (f32)
  gemm_bt_f32out<<<dim3(8, 16), 256, 0, stream>>>(attnB, WoB, out, 2048, 2048, 2048);
}
